// ODERNN_53987738911257
// MI455X (gfx1250) — matrix operand order repaired and checked by reading, not compiled, not run on hardware
//
#include <hip/hip_runtime.h>

// ---------------------------------------------------------------------------
// ODE-RNN on MI455X (gfx1250): all GEMMs via v_wmma_f32_16x16x32_bf16
// (bf16 inputs, fp32 accumulate), double-buffered LDS pipeline, async
// global->LDS staging (GLOBAL_LOAD_ASYNC_TO_LDS_B128) for bf16 weight tiles.
// ---------------------------------------------------------------------------

typedef __attribute__((ext_vector_type(16))) __bf16 bf16x16;
typedef __attribute__((ext_vector_type(8)))  __bf16 bf16x8;
typedef __attribute__((ext_vector_type(4)))  __bf16 bf16x4;
typedef __attribute__((ext_vector_type(8)))  float  f32x8;
typedef __attribute__((ext_vector_type(4)))  float  f32x4;
typedef int v4i_gcc __attribute__((vector_size(16)));   // builtin's pointee type

#define DT_STEP 0.05f

// ---- feature probes (compile-time safe) -----------------------------------
#if defined(__has_builtin)
#  if __has_builtin(__builtin_amdgcn_global_load_async_to_lds_b128)
#    define HAVE_ASYNC_COPY 1
#  endif
#endif
#ifndef HAVE_ASYNC_COPY
#  define HAVE_ASYNC_COPY 0
#endif

static __device__ __forceinline__ __bf16 f2bf(float f) {
    unsigned u = __builtin_bit_cast(unsigned, f);
    u = (u + 0x7FFFu + ((u >> 16) & 1u)) >> 16;   // round-to-nearest-even
    unsigned short s = (unsigned short)u;
    return __builtin_bit_cast(__bf16, s);
}

static __device__ __forceinline__ float fast_tanh(float x) {
#if defined(__has_builtin) && __has_builtin(__builtin_amdgcn_tanhf)
    return __builtin_amdgcn_tanhf(x);          // V_TANH_F32 trans op
#else
    return tanhf(x);
#endif
}

// 16-byte global(bf16) -> LDS chunk copy: async on CDNA5 when available.
static __device__ __forceinline__ void stage_b16(const __bf16* g, __bf16* l) {
#if HAVE_ASYNC_COPY
    typedef __attribute__((address_space(1))) v4i_gcc* gvec_p;   // global v4i*
    typedef __attribute__((address_space(3))) v4i_gcc* lvec_p;   // LDS v4i*
    gvec_p gp = (gvec_p)(unsigned long long)(size_t)g;  // reinterpret (drops const)
    lvec_p lp = (lvec_p)(v4i_gcc*)l;                    // reinterpret + addrspacecast
    __builtin_amdgcn_global_load_async_to_lds_b128(gp, lp, 0, 0);
#else
    *(bf16x8*)l = *(const bf16x8*)g;
#endif
}

static __device__ __forceinline__ void async_wait0() {
#if HAVE_ASYNC_COPY
#  if defined(__has_builtin) && __has_builtin(__builtin_amdgcn_s_wait_asynccnt)
    __builtin_amdgcn_s_wait_asynccnt(0);
#  else
    asm volatile("s_wait_asynccnt 0x0" ::: "memory");
#  endif
#endif
}

// -------------------------- small helper kernels ---------------------------

__global__ void cvt_f32_bf16_kernel(const float* __restrict__ src,
                                    __bf16* __restrict__ dst, int n) {
    int i = blockIdx.x * blockDim.x + threadIdx.x;
    if (i < n) dst[i] = f2bf(src[i]);
}

__global__ void zero_f32_kernel(float* __restrict__ p, int n) {
    for (int i = blockIdx.x * blockDim.x + threadIdx.x; i < n;
         i += gridDim.x * blockDim.x)
        p[i] = 0.0f;
}

// ----------------------- fused WMMA GEMM (bf16 -> f32) ---------------------
// out[orow, n] = act( alpha * (sum_k A[grow,k]*Bw[n,k] + bias[n]) + add[m,n] )
//   grow = rowidx ? rowidx[m] : m   (fused gather, hoisted out of K loop)
//   orow = outidx ? outidx[m] : m   (fused scatter)
// act: 0 = none, 1 = tanh, 2 = relu
// Requires M%128==0, N%128==0, K%32==0 (true for all shapes here).

#define BM 128
#define BN 128
#define BK 32

__global__ __launch_bounds__(256)
void gemm_bf16_wmma(const float* __restrict__ A, int lda,
                    const int* __restrict__ rowidx,
                    const __bf16* __restrict__ Bw, int ldb,     // Bw is [N,K]
                    const float* __restrict__ bias,
                    const float* __restrict__ addsrc,
                    float* __restrict__ C,
                    const int* __restrict__ outidx, int ldc,
                    int M, int N, int K, float alpha, int act) {
    // double-buffered tiles: 4 x 8KB = 32KB of the 320KB/WGP LDS
    __shared__ __bf16 As[2][BM][BK];
    __shared__ __bf16 Bs[2][BN][BK];
    const int ABUF = BM * BK;   // elements per A buffer
    const int BBUF = BN * BK;   // elements per B buffer

    const int tid   = threadIdx.x;
    const int wid   = tid >> 5;          // wave 0..7 (wave32)
    const int lane  = tid & 31;
    const int lhalf = lane >> 4;         // 0/1 (lane half)
    const int lrow  = lane & 15;
    const int wm    = wid & 3;           // wave row group: 32 rows each
    const int wn    = wid >> 2;          // wave col group: 64 cols each
    const int m0    = blockIdx.x * BM;
    const int n0    = blockIdx.y * BN;

    // ---- loop-invariant staging addresses (gather resolved once) ----------
    const float* aptr[4];                // global A row chunk base (k=0)
    __bf16*      adst[4];                // LDS dest in buffer 0
#pragma unroll
    for (int j = 0; j < 4; ++j) {
        int v  = tid + j * 256;          // 0..1023 float4 chunks
        int r  = v >> 3;                 // row 0..127
        int kq = (v & 7) << 2;           // 0,4,...,28
        int gr = rowidx ? rowidx[m0 + r] : (m0 + r);
        aptr[j] = A + (size_t)gr * lda + kq;
        adst[j] = &As[0][r][kq];
    }
    const __bf16* bptr[2];
    __bf16*       bdst[2];
#pragma unroll
    for (int j = 0; j < 2; ++j) {
        int c    = tid + j * 256;        // 0..511 16B chunks
        int nn   = c >> 2;               // 0..127
        int part = (c & 3) << 3;         // 0,8,16,24
        bptr[j] = Bw + (size_t)(n0 + nn) * ldb + part;
        bdst[j] = &Bs[0][nn][part];
    }

    f32x8 acc[2][4];
#pragma unroll
    for (int i = 0; i < 2; ++i)
#pragma unroll
        for (int j = 0; j < 4; ++j)
#pragma unroll
            for (int v = 0; v < 8; ++v) acc[i][j][v] = 0.0f;

    // ---- prologue: stage tile 0 into buffer 0 -----------------------------
    f32x4 areg[4];
#pragma unroll
    for (int j = 0; j < 2; ++j) stage_b16(bptr[j], bdst[j]);
#pragma unroll
    for (int j = 0; j < 4; ++j) areg[j] = *(const f32x4*)(aptr[j]);
#pragma unroll
    for (int j = 0; j < 4; ++j) {
        bf16x4 bv = { f2bf(areg[j].x), f2bf(areg[j].y),
                      f2bf(areg[j].z), f2bf(areg[j].w) };
        *(bf16x4*)adst[j] = bv;
    }
    async_wait0();
    __syncthreads();

    const int nk = K / BK;
    int cur = 0;
    for (int i = 0; i < nk; ++i) {
        const int nxt  = cur ^ 1;
        const bool more = (i + 1) < nk;

        // ---- issue next tile's global traffic before touching LDS --------
        if (more) {
            const int k0n = (i + 1) * BK;
#pragma unroll
            for (int j = 0; j < 2; ++j)
                stage_b16(bptr[j] + k0n, bdst[j] + nxt * BBUF);
#pragma unroll
            for (int j = 0; j < 4; ++j)
                areg[j] = *(const f32x4*)(aptr[j] + k0n);
            if (i + 2 < nk)
                __builtin_prefetch(aptr[tid & 3] + (i + 2) * BK, 0, 0);
        }

        // ---- fragments in contiguous k order (same for A and B), from buffer `cur` -------------
        // A 16x32 (16-bit): lane holds row=base+lrow,
        //   K = lhalf*16..+15
        bf16x16 afrag[2];
#pragma unroll
        for (int rt = 0; rt < 2; ++rt) {
            const __bf16* p = &As[cur][wm * 32 + rt * 16 + lrow][lhalf * 16];
            bf16x8 lo = *(const bf16x8*)p;
            bf16x8 hi = *(const bf16x8*)(p + 8);
#pragma unroll
            for (int q = 0; q < 8; ++q) {
                afrag[rt][q]     = lo[q];
                afrag[rt][8 + q] = hi[q];
            }
        }
        // B 32x16 (16-bit): lane holds col=base+lrow, K = lhalf*16..+15
        bf16x16 bfrag[4];
#pragma unroll
        for (int ct = 0; ct < 4; ++ct)
            bfrag[ct] = *(const bf16x16*)
                (&Bs[cur][wn * 64 + ct * 16 + lrow][lhalf * 16]);

        // ---- 8 WMMAs per wave per K-step ----------------------------------
#pragma unroll
        for (int rt = 0; rt < 2; ++rt)
#pragma unroll
            for (int ct = 0; ct < 4; ++ct)
                acc[rt][ct] = __builtin_amdgcn_wmma_f32_16x16x32_bf16(
                    false, afrag[rt], false, bfrag[ct],
                    (short)0, acc[rt][ct], false, false);

        // ---- finish staging next tile into buffer `nxt` -------------------
        if (more) {
#pragma unroll
            for (int j = 0; j < 4; ++j) {
                bf16x4 bv = { f2bf(areg[j].x), f2bf(areg[j].y),
                              f2bf(areg[j].z), f2bf(areg[j].w) };
                *(bf16x4*)(adst[j] + nxt * ABUF) = bv;
            }
            async_wait0();
        }
        __syncthreads();
        cur = nxt;
    }

    // ---- epilogue: C/D layout lane(L) VGPR(v) -> M = v + 8*(L/16), N = L%16
#pragma unroll
    for (int rt = 0; rt < 2; ++rt) {
        int mloc = wm * 32 + rt * 16 + lhalf * 8;
#pragma unroll
        for (int ct = 0; ct < 4; ++ct) {
            int col = n0 + wn * 64 + ct * 16 + lrow;
            float bv = bias ? bias[col] : 0.0f;
#pragma unroll
            for (int v = 0; v < 8; ++v) {
                int mrow = m0 + mloc + v;
                float x = (acc[rt][ct][v] + bv) * alpha;
                if (addsrc) x += addsrc[(size_t)mrow * ldc + col];
                if (act == 1)      x = fast_tanh(x);
                else if (act == 2) x = fmaxf(x, 0.0f);
                int orow = outidx ? outidx[mrow] : mrow;
                C[(size_t)orow * ldc + col] = x;
            }
        }
    }
}

// ------------------------------- loss kernel -------------------------------

__global__ void loss_mae_kernel(const float* __restrict__ X,
                                const float* __restrict__ Mk,
                                const float* __restrict__ P,
                                float* __restrict__ acc, int n) {
    __shared__ float sl[256];
    __shared__ float sm[256];
    float l = 0.0f, m = 0.0f;
    for (int i = blockIdx.x * blockDim.x + threadIdx.x; i < n;
         i += gridDim.x * blockDim.x) {
        float mk = Mk[i];
        l += fabsf(X[i] - P[i]) * mk;
        m += mk;
    }
    sl[threadIdx.x] = l;
    sm[threadIdx.x] = m;
    __syncthreads();
    for (int s = 128; s > 0; s >>= 1) {
        if (threadIdx.x < (unsigned)s) {
            sl[threadIdx.x] += sl[threadIdx.x + s];
            sm[threadIdx.x] += sm[threadIdx.x + s];
        }
        __syncthreads();
    }
    if (threadIdx.x == 0) {
        atomicAdd(&acc[0], sl[0]);
        atomicAdd(&acc[1], sm[0]);
    }
}

__global__ void finalize_kernel(const float* __restrict__ acc,
                                float* __restrict__ out) {
    out[0] = acc[0];
    out[1] = acc[0] / acc[1];
}

// --------------------------------- driver ----------------------------------

extern "C" void kernel_launch(void* const* d_in, const int* in_sizes, int n_in,
                              void* d_out, int out_size, void* d_ws,
                              size_t ws_size, hipStream_t stream) {
    (void)in_sizes; (void)n_in; (void)out_size; (void)ws_size;

    const float* X    = (const float*)d_in[0];
    const float* Mm   = (const float*)d_in[1];
    const int*   bidx = (const int*)d_in[2];
    const float* W_ih = (const float*)d_in[3];
    const float* b_ih = (const float*)d_in[4];
    const float* W_hh = (const float*)d_in[5];
    const float* b_hh = (const float*)d_in[6];
    const float* Wo1  = (const float*)d_in[7];
    const float* bo1  = (const float*)d_in[8];
    const float* Wo2  = (const float*)d_in[9];
    const float* bo2  = (const float*)d_in[10];
    const float* Wp1  = (const float*)d_in[11];
    const float* bp1  = (const float*)d_in[12];
    const float* Wp2  = (const float*)d_in[13];
    const float* bp2  = (const float*)d_in[14];

    const int NS = 16384, H = 512, D = 128, KOBS = 40, E = 4096;

    // workspace layout
    char*   ws    = (char*)d_ws;
    float*  h0    = (float*)ws;                                   // 32 MB
    float*  h1    = (float*)(ws + ((size_t)32 << 20));            // 32 MB
    float*  tbig  = (float*)(ws + ((size_t)64 << 20));            // 32 MB
    float*  t2    = tbig;                                         // [E,H] reuse
    float*  t3    = (float*)(ws + ((size_t)72 << 20));            // [E,H]
    float*  pbuf  = (float*)(ws + ((size_t)80 << 20));            // [E,D]
    float*  accum = (float*)(ws + ((size_t)96 << 20));            // 2 floats
    __bf16* wihb  = (__bf16*)(ws + ((size_t)96 << 20) + 256);     // [H,D]
    __bf16* whhb  = wihb + (size_t)H * D;                         // [H,H]
    __bf16* wo1b  = whhb + (size_t)H * H;
    __bf16* wo2b  = wo1b + (size_t)H * H;
    __bf16* wp1b  = wo2b + (size_t)H * H;
    __bf16* wp2b  = wp1b + (size_t)H * H;                         // [D,H]

    // fp32 -> bf16 weight conversion (layout preserved: [N,K] = torch layout)
    cvt_f32_bf16_kernel<<<(H * D + 255) / 256, 256, 0, stream>>>(W_ih, wihb, H * D);
    cvt_f32_bf16_kernel<<<(H * H + 255) / 256, 256, 0, stream>>>(W_hh, whhb, H * H);
    cvt_f32_bf16_kernel<<<(H * H + 255) / 256, 256, 0, stream>>>(Wo1,  wo1b, H * H);
    cvt_f32_bf16_kernel<<<(H * H + 255) / 256, 256, 0, stream>>>(Wo2,  wo2b, H * H);
    cvt_f32_bf16_kernel<<<(H * H + 255) / 256, 256, 0, stream>>>(Wp1,  wp1b, H * H);
    cvt_f32_bf16_kernel<<<(D * H + 255) / 256, 256, 0, stream>>>(Wp2,  wp2b, D * H);

    zero_f32_kernel<<<2048, 256, 0, stream>>>(h0, NS * H);
    zero_f32_kernel<<<1, 32, 0, stream>>>(accum, 2);

    float* h[2] = { h0, h1 };
    int cur = 0;

    dim3 gBig(NS / BM, H / BN);   // 128 x 4 blocks
    dim3 gObs(E / BM, H / BN);    //  32 x 4
    dim3 gP2(E / BM, D / BN);     //  32 x 1

    for (int k = 0; k < KOBS; ++k) {
        const float* Xk  = X    + (size_t)k * E * D;
        const float* Mk  = Mm   + (size_t)k * E * D;
        const int*   idx = bidx + (size_t)k * E;

        // ---- 2 Euler steps: h <- h + DT*(tanh(h@Wo1^T+bo1)@Wo2^T + bo2)
        for (int s = 0; s < 2; ++s) {
            gemm_bf16_wmma<<<gBig, 256, 0, stream>>>(
                h[cur], H, nullptr, wo1b, H, bo1, nullptr,
                tbig, nullptr, H, NS, H, H, 1.0f, /*tanh*/1);
            gemm_bf16_wmma<<<gBig, 256, 0, stream>>>(
                tbig, H, nullptr, wo2b, H, bo2, h[cur],
                h[cur ^ 1], nullptr, H, NS, H, H, DT_STEP, /*none*/0);
            cur ^= 1;
        }

        // ---- p_model: relu(h[idx]@Wp1^T+bp1)@Wp2^T + bp2   (fused gather)
        gemm_bf16_wmma<<<gObs, 256, 0, stream>>>(
            h[cur], H, idx, wp1b, H, bp1, nullptr,
            t2, nullptr, H, E, H, H, 1.0f, /*relu*/2);
        gemm_bf16_wmma<<<gP2, 256, 0, stream>>>(
            t2, H, nullptr, wp2b, H, bp2, nullptr,
            pbuf, nullptr, D, E, D, H, 1.0f, /*none*/0);

        // ---- masked MAE accumulation
        loss_mae_kernel<<<512, 256, 0, stream>>>(Xk, Mk, pbuf, accum, E * D);

        // ---- RNNCell: h[idx] <- tanh(X@W_ih^T+b_ih + h[idx]@W_hh^T+b_hh)
        gemm_bf16_wmma<<<gObs, 256, 0, stream>>>(
            h[cur], H, idx, whhb, H, b_hh, nullptr,
            t3, nullptr, H, E, H, H, 1.0f, /*none*/0);
        gemm_bf16_wmma<<<gObs, 256, 0, stream>>>(
            Xk, D, nullptr, wihb, D, b_ih, t3,
            h[cur], idx, H, E, H, D, 1.0f, /*tanh*/1);   // fused scatter
    }

    // EXTRA_STEPS Euler props do not affect the returned losses -> skipped.
    finalize_kernel<<<1, 1, 0, stream>>>(accum, (float*)d_out);
}
